// NeuralIFMessagePassing_84756884619734
// MI455X (gfx1250) — matrix operand order repaired and checked by reading, not compiled, not run on hardware
//
#include <hip/hip_runtime.h>
#include <math.h>

typedef _Float16 h2   __attribute__((ext_vector_type(2)));
typedef _Float16 h4   __attribute__((ext_vector_type(4)));
typedef _Float16 v16h __attribute__((ext_vector_type(16)));
typedef float    v8f  __attribute__((ext_vector_type(8)));

#define NN 100000      // nodes
#define NE 1600000     // edges
#define KE 224         // padded edge-MLP input dim (208 -> 224)
#define KE_REAL 208
#define HID 128
#define OUTD 64

// ---- workspace layout (bytes) ----
#define AGG_OFF   0            // float [NN*64]  = 25,600,000 B
#define W1TE_OFF  25600000     // f16 [128][224] = 57,344 B  (edge w1^T, K padded)
#define W2TE_OFF  25657344     // f16 [64][128]  = 16,384 B  (edge w2^T)
#define W1TN_OFF  25673728     // f16 [128][128] = 32,768 B  (node w1^T)
#define W2TN_OFF  25706496     // f16 [64][128]  = 16,384 B  (node w2^T)

// Branch-free erf-GELU (Abramowitz-Stegun 7.1.26, |err| < 1.5e-7):
// ~12 VALU ops incl. v_rcp_f32 + v_exp_f32, vs ~30+ for libm erff.
__device__ __forceinline__ float gelu_erf(float x) {
    float a = __builtin_fabsf(x) * 0.70710678118654752f;
    float t = __builtin_amdgcn_rcpf(__builtin_fmaf(0.3275911f, a, 1.0f));
    float p = t * __builtin_fmaf(t,
                  __builtin_fmaf(t,
                    __builtin_fmaf(t,
                      __builtin_fmaf(t, 1.061405429f, -1.453152027f),
                      1.421413741f),
                    -0.284496736f),
                  0.254829592f);
    float e = __builtin_amdgcn_exp2f(-a * a * 1.4426950408889634f);
    float erfa = __builtin_fmaf(-p, e, 1.0f);
    float erfv = __builtin_copysignf(erfa, x);
    return 0.5f * x * (1.0f + erfv);
}

// A-matrix fragment, 16-bit 16x32 layout (ISA 7.12.2):
// lane holds row M = lane&15; VGPR j (j=0..7): K = kc + (j>>2)*16 + half*8 + (j&3)*2, +1
__device__ __forceinline__ v16h frag_a(const _Float16* rowp, int kc, int half) {
    v16h f;
#pragma unroll
    for (int j = 0; j < 8; ++j) {
        int k = kc + ((j >> 2) << 4) + (half << 3) + ((j & 3) << 1);
        h2 v = *(const h2*)(rowp + k);
        f[2 * j] = v.x;
        f[2 * j + 1] = v.y;
    }
    return f;
}

// B-matrix fragment, 16-bit 32x16: column n=lane&15; VGPR j: K = kc + (j>>2)*16 + half*8 + (j&3)*2, +1
// (same K order as frag_a; W^T storage).
__device__ __forceinline__ v16h frag_b(const _Float16* colp, int kc, int half) {
    v16h f;
#pragma unroll
    for (int j = 0; j < 8; ++j) {
        int k = kc + ((j >> 2) << 4) + (half << 3) + ((j & 3) << 1);
        h2 v = *(const h2*)(colp + k);
        f[2 * j] = v.x;
        f[2 * j + 1] = v.y;
    }
    return f;
}

// ---------------- prep: zero agg + transpose/convert weights to f16 ----------------
__global__ __launch_bounds__(256) void prep_kernel(
    const float* __restrict__ ew1, const float* __restrict__ ew2,
    const float* __restrict__ nw1, const float* __restrict__ nw2,
    _Float16* __restrict__ w1te, _Float16* __restrict__ w2te,
    _Float16* __restrict__ w1tn, _Float16* __restrict__ w2tn,
    float* __restrict__ agg)
{
    int t = blockIdx.x * 256 + threadIdx.x;
    if (t < NN * OUTD) { agg[t] = 0.0f; return; }
    t -= NN * OUTD;
    if (t < HID * KE) {                       // edge w1^T [128][224], pad K
        int n = t / KE, k = t % KE;
        w1te[t] = (k < KE_REAL) ? (_Float16)ew1[k * HID + n] : (_Float16)0.0f;
        return;
    }
    t -= HID * KE;
    if (t < OUTD * HID) {                     // edge w2^T [64][128]
        int n = t / HID, k = t % HID;
        w2te[t] = (_Float16)ew2[k * OUTD + n];
        return;
    }
    t -= OUTD * HID;
    if (t < HID * HID) {                      // node w1^T [128][128]
        int n = t / HID, k = t % HID;
        w1tn[t] = (_Float16)nw1[k * HID + n];
        return;
    }
    t -= HID * HID;
    {                                         // node w2^T [64][128]
        int n = t / HID, k = t % HID;
        w2tn[t] = (_Float16)nw2[k * OUTD + n];
    }
}

// ---------------- edge MLP + scatter-add ----------------
// 1 wave = 16 edges; 8 waves / block; NE/16/8 = 12,500 blocks exactly.
__global__ __launch_bounds__(256) void edge_kernel(
    const float* __restrict__ node_x, const int* __restrict__ edge_index,
    const float* __restrict__ edge_latent, const float* __restrict__ edge_skip,
    const _Float16* __restrict__ w1t, const float* __restrict__ b1,
    const _Float16* __restrict__ w2t, const float* __restrict__ b2,
    float* __restrict__ e_out, float* __restrict__ agg)
{
    __shared__ _Float16 srow[8][16 * KE];   // 57,344 B (reused as h, then as f32 e-tile)
    __shared__ int sdst[8][16];

    const int w    = threadIdx.x >> 5;
    const int lane = threadIdx.x & 31;
    const int m    = lane & 15;   // row within tile / column within N-tile
    const int half = lane >> 4;
    const long tile  = (long)blockIdx.x * 8 + w;
    const long ebase = tile * 16;

    _Float16* R = srow[w];

    // --- gather: build 16 rows of [node_x[src] | node_x[dst] | latent | skip | pad] in f16 ---
    int idx = edge_index[(long)half * NE + ebase + m];  // half==0 -> src, half==1 -> dst
    if (half) sdst[w][m] = idx;
    {
        const float4* sv = (const float4*)(node_x + (long)idx * 64);
        _Float16* dp = R + m * KE + half * 64;
#pragma unroll
        for (int j = 0; j < 16; ++j) {
            float4 v = sv[j];
            h4 h = {(_Float16)v.x, (_Float16)v.y, (_Float16)v.z, (_Float16)v.w};
            *(h4*)(dp + j * 4) = h;
        }
    }
    if (half == 0) {
        const float4* lv = (const float4*)(edge_latent + (ebase + m) * 64);
        _Float16* dp = R + m * KE + 128;
#pragma unroll
        for (int j = 0; j < 16; ++j) {
            float4 v = lv[j];
            h4 h = {(_Float16)v.x, (_Float16)v.y, (_Float16)v.z, (_Float16)v.w};
            *(h4*)(dp + j * 4) = h;
        }
    } else {
        const float4* sv = (const float4*)(edge_skip + (ebase + m) * 16);
        _Float16* dp = R + m * KE + 192;
#pragma unroll
        for (int j = 0; j < 4; ++j) {
            float4 v = sv[j];
            h4 h = {(_Float16)v.x, (_Float16)v.y, (_Float16)v.z, (_Float16)v.w};
            *(h4*)(dp + j * 4) = h;
        }
        h4 z = {};
#pragma unroll
        for (int j = 0; j < 4; ++j) *(h4*)(dp + 16 + j * 4) = z;
    }

    // --- GEMM1: [16 x 224] @ [224 x 128] -> acc1 (8 N-tiles of 16x16 f32) ---
    // Batch-load all 8 B fragments per K-chunk (one clause / one wait), then 8 WMMAs.
    const _Float16* arow = R + m * KE;
    v8f acc1[8];
#pragma unroll
    for (int t = 0; t < 8; ++t) { v8f z = {}; acc1[t] = z; }
#pragma unroll
    for (int kc = 0; kc < KE; kc += 32) {
        v16h a = frag_a(arow, kc, half);
        v16h bfr[8];
#pragma unroll
        for (int t = 0; t < 8; ++t)
            bfr[t] = frag_b(w1t + (t * 16 + m) * KE, kc, half);
#pragma unroll
        for (int t = 0; t < 8; ++t)
            acc1[t] = __builtin_amdgcn_wmma_f32_16x16x32_f16(
                false, a, false, bfr[t], (short)0, acc1[t], false, false);
    }

    // --- bias + GELU, write h[16][128] f16 back into (consumed) LDS ---
#pragma unroll
    for (int t = 0; t < 8; ++t) {
        float bv = b1[t * 16 + m];
#pragma unroll
        for (int r = 0; r < 8; ++r) {
            float v = gelu_erf(acc1[t][r] + bv);
            R[(r + half * 8) * HID + t * 16 + m] = (_Float16)v;
        }
    }

    // --- GEMM2: [16 x 128] @ [128 x 64] -> e ---
    const _Float16* arow2 = R + m * HID;
    v8f acc2[4];
#pragma unroll
    for (int t = 0; t < 4; ++t) { v8f z = {}; acc2[t] = z; }
#pragma unroll
    for (int kc = 0; kc < HID; kc += 32) {
        v16h a = frag_a(arow2, kc, half);
        v16h bfr[4];
#pragma unroll
        for (int t = 0; t < 4; ++t)
            bfr[t] = frag_b(w2t + (t * 16 + m) * HID, kc, half);
#pragma unroll
        for (int t = 0; t < 4; ++t)
            acc2[t] = __builtin_amdgcn_wmma_f32_16x16x32_f16(
                false, a, false, bfr[t], (short)0, acc2[t], false, false);
    }

    // --- epilogue: bias, scatter-add into agg[dst], stage e tile in LDS (f32) ---
    float* EF = (float*)R;    // 16x64 f32 = 4096 B, aliases fully-consumed h region
#pragma unroll
    for (int t = 0; t < 4; ++t) {
        float bv = b2[t * 16 + m];
#pragma unroll
        for (int r = 0; r < 8; ++r) {
            int mrow = r + half * 8;
            float v = acc2[t][r] + bv;
            EF[mrow * OUTD + t * 16 + m] = v;
            atomicAdd(&agg[(long)sdst[w][mrow] * OUTD + t * 16 + m], v);
        }
    }

    // --- coalesced e_out stores: lane covers half a row (32 floats = 8 float4) ---
    {
        int row = lane >> 1, part = lane & 1;
        const float4* sp = (const float4*)(EF + row * OUTD + part * 32);
        float4* dp = (float4*)(e_out + (ebase + row) * OUTD + part * 32);
#pragma unroll
        for (int j = 0; j < 8; ++j) dp[j] = sp[j];
    }
}

// ---------------- node MLP ----------------
// 1 wave = 16 nodes; 8 waves/block; ceil(6250/8)=782 blocks.
__global__ __launch_bounds__(256) void node_kernel(
    const float* __restrict__ node_x, const float* __restrict__ agg,
    const _Float16* __restrict__ w1t, const float* __restrict__ b1,
    const _Float16* __restrict__ w2t, const float* __restrict__ b2,
    float* __restrict__ x_out)
{
    __shared__ _Float16 srow[8][16 * HID];  // 32,768 B

    const int w    = threadIdx.x >> 5;
    const int lane = threadIdx.x & 31;
    const int m    = lane & 15;
    const int half = lane >> 4;
    const long tile  = (long)blockIdx.x * 8 + w;
    const long nbase = tile * 16;
    if (nbase >= NN) return;   // no block barriers used anywhere

    _Float16* R = srow[w];

    // rows = [node_x[i] | agg[i]] in f16
    {
        const float* src = (half == 0 ? node_x : agg) + (nbase + m) * 64;
        const float4* sv = (const float4*)src;
        _Float16* dp = R + m * HID + half * 64;
#pragma unroll
        for (int j = 0; j < 16; ++j) {
            float4 v = sv[j];
            h4 h = {(_Float16)v.x, (_Float16)v.y, (_Float16)v.z, (_Float16)v.w};
            *(h4*)(dp + j * 4) = h;
        }
    }

    // GEMM1: [16 x 128] @ [128 x 128]
    const _Float16* arow = R + m * HID;
    v8f acc1[8];
#pragma unroll
    for (int t = 0; t < 8; ++t) { v8f z = {}; acc1[t] = z; }
#pragma unroll
    for (int kc = 0; kc < HID; kc += 32) {
        v16h a = frag_a(arow, kc, half);
        v16h bfr[8];
#pragma unroll
        for (int t = 0; t < 8; ++t)
            bfr[t] = frag_b(w1t + (t * 16 + m) * HID, kc, half);
#pragma unroll
        for (int t = 0; t < 8; ++t)
            acc1[t] = __builtin_amdgcn_wmma_f32_16x16x32_f16(
                false, a, false, bfr[t], (short)0, acc1[t], false, false);
    }

    // bias + GELU -> h back into LDS (rows fully consumed)
#pragma unroll
    for (int t = 0; t < 8; ++t) {
        float bv = b1[t * 16 + m];
#pragma unroll
        for (int r = 0; r < 8; ++r) {
            float v = gelu_erf(acc1[t][r] + bv);
            R[(r + half * 8) * HID + t * 16 + m] = (_Float16)v;
        }
    }

    // GEMM2: [16 x 128] @ [128 x 64]
    v8f acc2[4];
#pragma unroll
    for (int t = 0; t < 4; ++t) { v8f z = {}; acc2[t] = z; }
#pragma unroll
    for (int kc = 0; kc < HID; kc += 32) {
        v16h a = frag_a(arow, kc, half);
        v16h bfr[4];
#pragma unroll
        for (int t = 0; t < 4; ++t)
            bfr[t] = frag_b(w2t + (t * 16 + m) * HID, kc, half);
#pragma unroll
        for (int t = 0; t < 4; ++t)
            acc2[t] = __builtin_amdgcn_wmma_f32_16x16x32_f16(
                false, a, false, bfr[t], (short)0, acc2[t], false, false);
    }

    // stage x tile in LDS (f32, aliases consumed h region: 16x64 f32 = 4096 B)
    float* XF = (float*)R;
#pragma unroll
    for (int t = 0; t < 4; ++t) {
        float bv = b2[t * 16 + m];
#pragma unroll
        for (int r = 0; r < 8; ++r) {
            int mrow = r + half * 8;
            XF[mrow * OUTD + t * 16 + m] = acc2[t][r] + bv;
        }
    }

    // coalesced x_out stores
    {
        int row = lane >> 1, part = lane & 1;
        const float4* sp = (const float4*)(XF + row * OUTD + part * 32);
        float4* dp = (float4*)(x_out + (nbase + row) * OUTD + part * 32);
#pragma unroll
        for (int j = 0; j < 8; ++j) dp[j] = sp[j];
    }
}

extern "C" void kernel_launch(void* const* d_in, const int* in_sizes, int n_in,
                              void* d_out, int out_size, void* d_ws, size_t ws_size,
                              hipStream_t stream) {
    const float* node_x      = (const float*)d_in[0];
    const int*   edge_index  = (const int*)d_in[1];
    const float* edge_latent = (const float*)d_in[2];
    const float* edge_skip   = (const float*)d_in[3];
    const float* ew1 = (const float*)d_in[4];
    const float* eb1 = (const float*)d_in[5];
    const float* ew2 = (const float*)d_in[6];
    const float* eb2 = (const float*)d_in[7];
    const float* nw1 = (const float*)d_in[8];
    const float* nb1 = (const float*)d_in[9];
    const float* nw2 = (const float*)d_in[10];
    const float* nb2 = (const float*)d_in[11];

    char* ws = (char*)d_ws;
    float*     agg  = (float*)(ws + AGG_OFF);
    _Float16*  w1te = (_Float16*)(ws + W1TE_OFF);
    _Float16*  w2te = (_Float16*)(ws + W2TE_OFF);
    _Float16*  w1tn = (_Float16*)(ws + W1TN_OFF);
    _Float16*  w2tn = (_Float16*)(ws + W2TN_OFF);

    float* x_out = (float*)d_out;               // [NN, 64]
    float* e_out = x_out + (long)NN * OUTD;     // [NE, 64]

    // 6,400,000 + 28,672 + 8,192 + 16,384 + 8,192 = 6,461,440 = 25,240 * 256
    prep_kernel<<<25240, 256, 0, stream>>>(ew1, ew2, nw1, nw2,
                                           w1te, w2te, w1tn, w2tn, agg);
    edge_kernel<<<12500, 256, 0, stream>>>(node_x, edge_index, edge_latent, edge_skip,
                                           w1te, eb1, w2te, eb2, e_out, agg);
    node_kernel<<<782, 256, 0, stream>>>(node_x, agg, w1tn, nb1, w2tn, nb2, x_out);
}
